// DisjointDecoderAE_27393301413996
// MI455X (gfx1250) — compile-verified, matrix operand order checked by reading, not run on hardware
//
#include <hip/hip_runtime.h>
#include <hip/hip_bf16.h>

#define B_DIM 2048
#define U_DIM 512
#define L_DIM 32
#define H_DIM 64
#define MT    128     // batch rows per tile (8 waves x 16 rows)
#define TPB   256
#define HS    66      // padded LDS row stride (elements) for 64-wide buffers
#define W1S   34      // padded LDS row stride for 32-wide w1T rows

typedef __attribute__((ext_vector_type(16))) __bf16       v16bf;
typedef __attribute__((ext_vector_type(8)))  float        v8f;
typedef __attribute__((ext_vector_type(8)))  unsigned int v8u;

union VUB { v8u u; v16bf b; };

__device__ __forceinline__ unsigned short f2bf(float f) {
  unsigned int x = __builtin_bit_cast(unsigned int, f);
  x += 0x7FFFu + ((x >> 16) & 1u);          // round to nearest even
  return (unsigned short)(x >> 16);
}
__device__ __forceinline__ float bf2f(unsigned short h) {
  unsigned int x = ((unsigned int)h) << 16;
  return __builtin_bit_cast(float, x);
}

// A fragment: 16x32 bf16 tile at (m0,k0) from row-major LDS buffer (stride even).
// Lane 0-15: M=lane, K-halves {0..7,16..23}; lane 16-31: same M rows, K-halves {8..15,24..31}.
__device__ __forceinline__ v16bf lds_load_afrag(const unsigned short* buf, int stride,
                                                int m0, int k0) {
  const int lane = threadIdx.x & 31;
  const int m  = m0 + (lane & 15);
  const int kh = (lane >> 4) & 1;
  const unsigned short* p = buf + m * stride + k0 + kh * 8;
  VUB r;
#pragma unroll
  for (int v = 0; v < 4; ++v) {
    r.u[v]     = *(const unsigned int*)(p + 2 * v);        // K = kh*8 + 2v, +1
    r.u[v + 4] = *(const unsigned int*)(p + 16 + 2 * v);   // K = 16 + kh*8 + 2v, +1
  }
  return r.b;
}

// B fragment: 32x16 bf16 tile at (k0,n0) from *transposed* (N-major) LDS weights.
__device__ __forceinline__ v16bf lds_load_bfrag(const unsigned short* wT, int stride,
                                                int k0, int n0) {
  const int lane = threadIdx.x & 31;
  const int n  = n0 + (lane & 15);
  const int kh = (lane >> 4) & 1;
  const unsigned short* p = wT + n * stride + k0 + kh * 8;
  VUB r;
#pragma unroll
  for (int v = 0; v < 4; ++v) {
    r.u[v]     = *(const unsigned int*)(p + 2 * v);
    r.u[v + 4] = *(const unsigned int*)(p + 16 + 2 * v);
  }
  return r.b;
}

__device__ __forceinline__ v8f wmma_bf16(v16bf a, v16bf b, v8f c) {
  return __builtin_amdgcn_wmma_f32_16x16x32_bf16(false, a, false, b, (short)0, c,
                                                 false, false);
}

// C fragment (16x16 f32) + bias (+ReLU) -> bf16 row-major LDS buffer.
// VGPR v: M = v (lanes 0-15) / v+8 (lanes 16-31); N = lane&15.
__device__ __forceinline__ void store_c_act(unsigned short* buf, int stride,
                                            int m0, int n0, v8f c,
                                            const float* bias, bool relu) {
  const int lane = threadIdx.x & 31;
  const int n  = n0 + (lane & 15);
  const int mb = m0 + ((lane >> 4) & 1) * 8;
  const float bv = bias[n];
#pragma unroll
  for (int v = 0; v < 8; ++v) {
    float f = c[v] + bv;
    if (relu) f = fmaxf(f, 0.0f);
    buf[(mb + v) * stride + n] = f2bf(f);
  }
}

// ---------------- shared encoder: x[B,512] -> z[B,32] ----------------
__global__ __launch_bounds__(TPB) void dae_encoder(
    const float* __restrict__ x,
    const float* __restrict__ We1, const float* __restrict__ be1,
    const float* __restrict__ We2, const float* __restrict__ be2,
    const float* __restrict__ We3, const float* __restrict__ be3,
    const float* __restrict__ We4, const float* __restrict__ be4,
    float* __restrict__ z) {
  __shared__ unsigned short bufA[MT * HS];
  __shared__ unsigned short bufB[MT * HS];
  __shared__ unsigned short wT[H_DIM * HS];
  __shared__ float bias[H_DIM];

  const int tid  = threadIdx.x;
  const int m0   = (tid >> 5) * 16;
  const int row0 = blockIdx.x * MT;

  // layer 1: [128,512] x [512,64], K chunked by 64 through LDS
  v8f acc[4] = {};
  for (int kc = 0; kc < U_DIM / H_DIM; ++kc) {
    for (int idx = tid; idx < MT * H_DIM; idx += TPB) {
      int r = idx >> 6, c = idx & 63;
      bufA[r * HS + c] = f2bf(x[(size_t)(row0 + r) * U_DIM + kc * 64 + c]);
    }
    for (int idx = tid; idx < H_DIM * H_DIM; idx += TPB) {
      int k = idx >> 6, n = idx & 63;
      wT[n * HS + k] = f2bf(We1[(size_t)(kc * 64 + k) * H_DIM + n]);
    }
    __syncthreads();
#pragma unroll
    for (int nt = 0; nt < 4; ++nt)
#pragma unroll
      for (int kt = 0; kt < 2; ++kt)
        acc[nt] = wmma_bf16(lds_load_afrag(bufA, HS, m0, kt * 32),
                            lds_load_bfrag(wT, HS, kt * 32, nt * 16), acc[nt]);
    __syncthreads();
  }
  for (int idx = tid; idx < H_DIM; idx += TPB) bias[idx] = be1[idx];
  __syncthreads();
#pragma unroll
  for (int nt = 0; nt < 4; ++nt) store_c_act(bufB, HS, m0, nt * 16, acc[nt], bias, true);

  // layer 2: bufB -> bufA
  for (int idx = tid; idx < H_DIM * H_DIM; idx += TPB) {
    int k = idx >> 6, n = idx & 63;
    wT[n * HS + k] = f2bf(We2[idx]);
  }
  for (int idx = tid; idx < H_DIM; idx += TPB) bias[idx] = be2[idx];
  __syncthreads();
  {
    v8f a2[4] = {};
#pragma unroll
    for (int nt = 0; nt < 4; ++nt)
#pragma unroll
      for (int kt = 0; kt < 2; ++kt)
        a2[nt] = wmma_bf16(lds_load_afrag(bufB, HS, m0, kt * 32),
                           lds_load_bfrag(wT, HS, kt * 32, nt * 16), a2[nt]);
#pragma unroll
    for (int nt = 0; nt < 4; ++nt) store_c_act(bufA, HS, m0, nt * 16, a2[nt], bias, true);
  }
  __syncthreads();

  // layer 3: bufA -> bufB
  for (int idx = tid; idx < H_DIM * H_DIM; idx += TPB) {
    int k = idx >> 6, n = idx & 63;
    wT[n * HS + k] = f2bf(We3[idx]);
  }
  for (int idx = tid; idx < H_DIM; idx += TPB) bias[idx] = be3[idx];
  __syncthreads();
  {
    v8f a3[4] = {};
#pragma unroll
    for (int nt = 0; nt < 4; ++nt)
#pragma unroll
      for (int kt = 0; kt < 2; ++kt)
        a3[nt] = wmma_bf16(lds_load_afrag(bufA, HS, m0, kt * 32),
                           lds_load_bfrag(wT, HS, kt * 32, nt * 16), a3[nt]);
#pragma unroll
    for (int nt = 0; nt < 4; ++nt) store_c_act(bufB, HS, m0, nt * 16, a3[nt], bias, true);
  }
  __syncthreads();

  // layer 4: [128,64] x [64,32] -> z (fp32, no relu)
  for (int idx = tid; idx < H_DIM * L_DIM; idx += TPB) {
    int k = idx >> 5, n = idx & 31;
    wT[n * HS + k] = f2bf(We4[idx]);
  }
  for (int idx = tid; idx < L_DIM; idx += TPB) bias[idx] = be4[idx];
  __syncthreads();
#pragma unroll
  for (int nt = 0; nt < 2; ++nt) {
    v8f c = {};
#pragma unroll
    for (int kt = 0; kt < 2; ++kt)
      c = wmma_bf16(lds_load_afrag(bufB, HS, m0, kt * 32),
                    lds_load_bfrag(wT, HS, kt * 32, nt * 16), c);
    const int lane = threadIdx.x & 31;
    const int n  = nt * 16 + (lane & 15);
    const int mb = m0 + ((lane >> 4) & 1) * 8;
    const float bv = bias[n];
#pragma unroll
    for (int v = 0; v < 8; ++v)
      z[(size_t)(row0 + mb + v) * L_DIM + n] = c[v] + bv;
  }
}

// ------------- disjoint decoders: one block per unit, fully fused -------------
__global__ __launch_bounds__(TPB) void dae_decoder(
    const float* __restrict__ z,
    const float* __restrict__ Wd1, const float* __restrict__ bd1,
    const float* __restrict__ Wd2, const float* __restrict__ bd2,
    const float* __restrict__ Wd3, const float* __restrict__ bd3,
    const float* __restrict__ Wd4, const float* __restrict__ bd4,
    float* __restrict__ out) {
  __shared__ unsigned short w1T[H_DIM * W1S];   // [n=64][k=32], N-major
  __shared__ unsigned short w2T[H_DIM * HS];    // [n=64][k=64]
  __shared__ unsigned short w3T[H_DIM * HS];
  __shared__ float w4[H_DIM];
  __shared__ float b1[H_DIM], b2[H_DIM], b3[H_DIM];
  __shared__ unsigned short bufA[MT * HS];
  __shared__ unsigned short bufB[MT * HS];

  const int tid = threadIdx.x;
  const int m0  = (tid >> 5) * 16;
  const int u   = blockIdx.x;

  // stage this unit's weights once (bf16, transposed)
  const float* W1 = Wd1 + (size_t)u * L_DIM * H_DIM;
  for (int idx = tid; idx < L_DIM * H_DIM; idx += TPB) {
    int l = idx >> 6, h = idx & 63;            // W1[l][h]
    w1T[h * W1S + l] = f2bf(W1[idx]);
  }
  const float* W2 = Wd2 + (size_t)u * H_DIM * H_DIM;
  const float* W3 = Wd3 + (size_t)u * H_DIM * H_DIM;
  for (int idx = tid; idx < H_DIM * H_DIM; idx += TPB) {
    int k = idx >> 6, n = idx & 63;
    w2T[n * HS + k] = f2bf(W2[idx]);
    w3T[n * HS + k] = f2bf(W3[idx]);
  }
  for (int idx = tid; idx < H_DIM; idx += TPB) {
    w4[idx] = Wd4[(size_t)u * H_DIM + idx];
    b1[idx] = bd1[(size_t)u * H_DIM + idx];
    b2[idx] = bd2[(size_t)u * H_DIM + idx];
    b3[idx] = bd3[(size_t)u * H_DIM + idx];
  }
  const float b4 = bd4[u];
  __syncthreads();

  for (int bt = 0; bt < B_DIM / MT; ++bt) {
    const int row0 = bt * MT;
    // z tile -> bufA (bf16)
    for (int idx = tid; idx < MT * L_DIM; idx += TPB) {
      int r = idx >> 5, c = idx & 31;
      bufA[r * HS + c] = f2bf(z[(size_t)(row0 + r) * L_DIM + c]);
    }
    __syncthreads();

    // layer 1: [128,32] x [32,64] -> bufB
#pragma unroll
    for (int nt = 0; nt < 4; ++nt) {
      v8f c = {};
      c = wmma_bf16(lds_load_afrag(bufA, HS, m0, 0),
                    lds_load_bfrag(w1T, W1S, 0, nt * 16), c);
      store_c_act(bufB, HS, m0, nt * 16, c, b1, true);
    }
    __syncthreads();

    // layer 2: bufB -> bufA
    {
      v8f a2[4] = {};
#pragma unroll
      for (int nt = 0; nt < 4; ++nt)
#pragma unroll
        for (int kt = 0; kt < 2; ++kt)
          a2[nt] = wmma_bf16(lds_load_afrag(bufB, HS, m0, kt * 32),
                             lds_load_bfrag(w2T, HS, kt * 32, nt * 16), a2[nt]);
#pragma unroll
      for (int nt = 0; nt < 4; ++nt) store_c_act(bufA, HS, m0, nt * 16, a2[nt], b2, true);
    }
    __syncthreads();

    // layer 3: bufA -> bufB
    {
      v8f a3[4] = {};
#pragma unroll
      for (int nt = 0; nt < 4; ++nt)
#pragma unroll
        for (int kt = 0; kt < 2; ++kt)
          a3[nt] = wmma_bf16(lds_load_afrag(bufA, HS, m0, kt * 32),
                             lds_load_bfrag(w3T, HS, kt * 32, nt * 16), a3[nt]);
#pragma unroll
      for (int nt = 0; nt < 4; ++nt) store_c_act(bufB, HS, m0, nt * 16, a3[nt], b3, true);
    }
    __syncthreads();

    // layer 4: per-row dot with w4 (fp32)
    if (tid < MT) {
      const unsigned short* hr = bufB + tid * HS;
      float s = b4;
#pragma unroll
      for (int h = 0; h < H_DIM; ++h) s += bf2f(hr[h]) * w4[h];
      out[(size_t)(row0 + tid) * U_DIM + u] = s;
    }
    __syncthreads();
  }
}

extern "C" void kernel_launch(void* const* d_in, const int* in_sizes, int n_in,
                              void* d_out, int out_size, void* d_ws, size_t ws_size,
                              hipStream_t stream) {
  (void)in_sizes; (void)n_in; (void)out_size; (void)ws_size;
  const float* x   = (const float*)d_in[0];
  const float* We1 = (const float*)d_in[1];
  const float* be1 = (const float*)d_in[2];
  const float* We2 = (const float*)d_in[3];
  const float* be2 = (const float*)d_in[4];
  const float* We3 = (const float*)d_in[5];
  const float* be3 = (const float*)d_in[6];
  const float* We4 = (const float*)d_in[7];
  const float* be4 = (const float*)d_in[8];
  const float* Wd1 = (const float*)d_in[9];
  const float* bd1 = (const float*)d_in[10];
  const float* Wd2 = (const float*)d_in[11];
  const float* bd2 = (const float*)d_in[12];
  const float* Wd3 = (const float*)d_in[13];
  const float* bd3 = (const float*)d_in[14];
  const float* Wd4 = (const float*)d_in[15];
  const float* bd4 = (const float*)d_in[16];

  float* z = (float*)d_ws;  // [B, L] fp32 latent scratch (256 KB)

  dae_encoder<<<B_DIM / MT, TPB, 0, stream>>>(x, We1, be1, We2, be2, We3, be3,
                                              We4, be4, z);
  dae_decoder<<<U_DIM, TPB, 0, stream>>>(z, Wd1, bd1, Wd2, bd2, Wd3, bd3,
                                         Wd4, bd4, (float*)d_out);
}
